// CoPEAttention_47854525612485
// MI455X (gfx1250) — compile-verified, matrix operand order checked by reading, not run on hardware
//
#include <hip/hip_runtime.h>
#include <hip/hip_fp16.h>
#include <math.h>

typedef _Float16 h16;
typedef __attribute__((ext_vector_type(16))) _Float16 v16h;
typedef __attribute__((ext_vector_type(8)))  _Float16 v8h;
typedef __attribute__((ext_vector_type(8)))  float    v8f;

#define CB 2
#define CT 1024
#define CD 1024
#define CH 16
#define CDH 64
#define CMAXPOS 64

// ---- CDNA5 async global->LDS copy (ASYNCcnt-tracked, no VGPR staging) ----
__device__ __forceinline__ void async_copy_b128(unsigned lds_off, const h16* g) {
  asm volatile("global_load_async_to_lds_b128 %0, %1, off"
               :: "v"(lds_off), "v"(g) : "memory");
}
__device__ __forceinline__ void wait_async_0() {
  asm volatile("s_wait_asynccnt 0" ::: "memory");
}

// ---------------- f32 -> f16 convert (grid-stride) ----------------
__global__ void cvt_f32_f16(const float* __restrict__ in, h16* __restrict__ out, long n) {
  long i = (long)blockIdx.x * blockDim.x + threadIdx.x;
  long stride = (long)gridDim.x * blockDim.x;
  for (; i < n; i += stride) out[i] = (h16)in[i];
}

// ---------------- V transpose: (B,T,H*DH) -> (B,H,DH,T), f16 ----------------
__global__ void transpose_V(const h16* __restrict__ V, h16* __restrict__ Vt) {
  __shared__ h16 tile[32][33];
  const int t0 = blockIdx.x * 32, d0 = blockIdx.y * 32, b = blockIdx.z;
  const int tx = threadIdx.x, ty = threadIdx.y;
  tile[ty][tx] = V[((long)b * CT + t0 + ty) * CD + d0 + tx];
  __syncthreads();
  const int dg = d0 + ty;                 // 64-wide heads, 32-wide tiles: no head crossing
  const int h = dg >> 6, dd = dg & 63;
  Vt[(((long)b * CH + h) * CDH + dd) * CT + (t0 + tx)] = tile[tx][ty];
}

// ---------------- batched WMMA GEMM: C[z] = A[z](MxK) * B[z](NxK)^T --------
// z offsets: base + (z/Hdiv)*s0 + (z%Hdiv)*s1 (expresses (b,h) batched views).
// Block = 128 threads (4 waves), tile 64x64, K-step 32, double-buffered LDS
// filled by async global->LDS b128 copies. Wave w owns M-strip
// [w*16,w*16+16) x all 64 N cols -> 4 WMMAs per K-step, A-frag reused x4.
__global__ void gemm_wmma_f16(
    const h16* __restrict__ A, long a_s0, long a_s1, int lda,
    const h16* __restrict__ Bm, long b_s0, long b_s1, int ldb,
    float* __restrict__ Cf, long c_s0, long c_s1, int ldc,
    h16* __restrict__ Ch, long ch_s0, long ch_s1, int ldch,
    const float* __restrict__ bias, int Hdiv, int M, int N, int Kd)
{
  __shared__ __align__(16) h16 As[2][64][40];
  __shared__ __align__(16) h16 Bs[2][64][40];

  const int tid  = threadIdx.x;          // 128 threads = 4 waves (wave32)
  const int lane = tid & 31;
  const int w    = tid >> 5;             // wave id 0..3: M-strip owner
  const int half = lane >> 4;            // lanes 0-15 / 16-31
  const int r    = lane & 15;

  const int  z  = blockIdx.z;
  const long zb = z / Hdiv, zh = z % Hdiv;
  const h16* Ab = A  + zb * a_s0 + zh * a_s1;
  const h16* Bb = Bm + zb * b_s0 + zh * b_s1;

  const int m0 = blockIdx.x * 64;
  const int n0 = blockIdx.y * 64;

  // staging map: tile is 64 rows x 32 halfs = 256 x 16B chunks; 2 per thread.
  const int row0 = tid >> 2, seg = (tid & 3) * 8;
  const int row1 = row0 + 32;

  // per-thread global sources and LDS destinations (byte offsets into LDS)
  const h16* gA0 = Ab + (long)(m0 + row0) * lda + seg;
  const h16* gA1 = Ab + (long)(m0 + row1) * lda + seg;
  const h16* gB0 = Bb + (long)(n0 + row0) * ldb + seg;
  const h16* gB1 = Bb + (long)(n0 + row1) * ldb + seg;
  const unsigned aoff0 = (unsigned)(size_t)(&As[0][row0][seg]);
  const unsigned aoff1 = (unsigned)(size_t)(&As[0][row1][seg]);
  const unsigned boff0 = (unsigned)(size_t)(&Bs[0][row0][seg]);
  const unsigned boff1 = (unsigned)(size_t)(&Bs[0][row1][seg]);
  const unsigned bufstride = (unsigned)sizeof(As[0]);   // 64*40*2 bytes

  // ---- prologue: async-stage K-tile 0 into buffer 0 ----
  async_copy_b128(aoff0, gA0);
  async_copy_b128(aoff1, gA1);
  async_copy_b128(boff0, gB0);
  async_copy_b128(boff1, gB1);
  wait_async_0();
  __syncthreads();

  v8f acc[4] = {};
  const int nIter = Kd >> 5;
  for (int it = 0; it < nIter; ++it) {
    const int cur = it & 1;
    const int kn  = (it + 1) << 5;
    if (it + 1 < nIter) {   // async-stage next tile into the other buffer
      const unsigned nb = (unsigned)(cur ^ 1) * bufstride;
      async_copy_b128(aoff0 + nb, gA0 + kn);
      async_copy_b128(aoff1 + nb, gA1 + kn);
      async_copy_b128(boff0 + nb, gB0 + kn);
      async_copy_b128(boff1 + nb, gB1 + kn);
      if (kn + 32 < Kd) {   // CDNA5 global_prefetch_b8 one tile further ahead
        __builtin_prefetch(gA0 + kn + 32, 0, 1);
        __builtin_prefetch(gB0 + kn + 32, 0, 1);
      }
    }

    // A fragment (reused across 4 N-tiles): lane r = row M; halves cover
    // K {0..7,16..23} / {8..15,24..31}
    const h16* ap = &As[cur][w * 16 + r][0];
    v8h a0 = *(const v8h*)(ap + half * 8);
    v8h a1 = *(const v8h*)(ap + 16 + half * 8);
    v16h av = __builtin_shufflevector(a0, a1, 0,1,2,3,4,5,6,7,8,9,10,11,12,13,14,15);

    // load all B fragments first so WMMAs issue back-to-back (one dscnt wait)
    v8h blo[4], bhi[4];
    #pragma unroll
    for (int nt = 0; nt < 4; ++nt) {
      const h16* bp = &Bs[cur][nt * 16 + r][0];
      blo[nt] = *(const v8h*)(bp + half * 8);
      bhi[nt] = *(const v8h*)(bp + 16 + half * 8);
    }
    #pragma unroll
    for (int nt = 0; nt < 4; ++nt) {
      v16h bv = __builtin_shufflevector(blo[nt], bhi[nt],
                                        0,1,2,3,4,5,6,7,8,9,10,11,12,13,14,15);
      acc[nt] = __builtin_amdgcn_wmma_f32_16x16x32_f16(false, av, false, bv,
                                                       (short)0, acc[nt], false, false);
    }

    if (it + 1 < nIter) {   // publish next buffer to all waves
      wait_async_0();
      __syncthreads();
    }
  }

  // ---- epilogue: D layout VGPR v -> row v (+8 for hi half), col = lane&15 ----
  float*  Cfb = Cf ? Cf + zb * c_s0  + zh * c_s1  : nullptr;
  h16*    Chb = Ch ? Ch + zb * ch_s0 + zh * ch_s1 : nullptr;
  #pragma unroll
  for (int nt = 0; nt < 4; ++nt) {
    const int n  = n0 + nt * 16 + r;
    const float bn = bias ? bias[n] : 0.0f;
    #pragma unroll
    for (int v = 0; v < 8; ++v) {
      int m = m0 + w * 16 + v + 8 * half;
      float val = acc[nt][v] + bn;
      if (Cfb) Cfb[(long)m * ldc  + n] = val;
      if (Chb) Chb[(long)m * ldch + n] = (h16)val;
    }
  }
}

// ---------------- CoPE gate-cumsum + interpolation + softmax ----------------
// One wave32 per (b,h,i) row. pos_ij = clip(total_i - cum_ij, 0, 63),
// scores = (qk + lerp(qpe_tab, pos)) / sqrt(dh), causal softmax -> f16 attn.
__global__ void cope_softmax(const float* __restrict__ qk,
                             const float* __restrict__ qpe_tab,
                             h16* __restrict__ attn)
{
  __shared__ float qkrow[CT];
  __shared__ float buf[CT];
  const int i    = blockIdx.x;
  const int bh   = blockIdx.y;
  const int lane = threadIdx.x;            // wave32
  const float inv_scale = 0.125f;          // 1/sqrt(64)

  const float* qrow = qk      + ((long)bh * CT + i) * CT;
  const float* tab  = qpe_tab + ((long)bh * CT + i) * CMAXPOS;
  h16*         arow = attn    + ((long)bh * CT + i) * CT;

  // pass 1: sigmoid gates + wave inclusive scan with running carry
  float carry = 0.0f;
  for (int j0 = 0; j0 < CT; j0 += 32) {
    int j = j0 + lane;
    float q = qrow[j];
    qkrow[j] = q;
    float v = (j <= i) ? 1.0f / (1.0f + __expf(-q * inv_scale)) : 0.0f;
    #pragma unroll
    for (int off = 1; off < 32; off <<= 1) {
      float nb = __shfl_up(v, off, 32);
      if (lane >= off) v += nb;
    }
    buf[j] = carry + v;
    carry += __shfl(v, 31, 32);
  }
  __syncthreads();
  const float total = buf[i];

  // pass 2: scores with interpolated q.pos_emb; track row max
  float mx = -INFINITY;
  for (int j0 = 0; j0 < CT; j0 += 32) {
    int j = j0 + lane;
    float s;
    if (j <= i) {
      float pos = fminf(fmaxf(total - buf[j], 0.0f), (float)(CMAXPOS - 1));
      float pf  = floorf(pos);
      float a   = pos - pf;
      int fi = (int)pf;
      int ci = min(fi + 1, CMAXPOS - 1);
      float qpe = (1.0f - a) * tab[fi] + a * tab[ci];
      s = (qkrow[j] + qpe) * inv_scale;
    } else {
      s = -INFINITY;
    }
    buf[j] = s;
    mx = fmaxf(mx, s);
  }
  #pragma unroll
  for (int off = 16; off >= 1; off >>= 1) mx = fmaxf(mx, __shfl_xor(mx, off, 32));

  // pass 3: exp/sum
  float sum = 0.0f;
  for (int j0 = 0; j0 < CT; j0 += 32) {
    int j = j0 + lane;
    float e = (j <= i) ? __expf(buf[j] - mx) : 0.0f;
    buf[j] = e;
    sum += e;
  }
  #pragma unroll
  for (int off = 16; off >= 1; off >>= 1) sum += __shfl_xor(sum, off, 32);
  const float inv = 1.0f / sum;

  for (int j0 = 0; j0 < CT; j0 += 32) {
    int j = j0 + lane;
    arow[j] = (h16)(buf[j] * inv);
  }
}

// ---------------- driver ----------------
extern "C" void kernel_launch(void* const* d_in, const int* in_sizes, int n_in,
                              void* d_out, int out_size, void* d_ws, size_t ws_size,
                              hipStream_t stream) {
  const float* x  = (const float*)d_in[0];
  const float* Wq = (const float*)d_in[1];
  const float* bq = (const float*)d_in[2];
  const float* Wk = (const float*)d_in[3];
  const float* bk = (const float*)d_in[4];
  const float* Wv = (const float*)d_in[5];
  const float* bv = (const float*)d_in[6];
  const float* Wo = (const float*)d_in[7];
  const float* bo = (const float*)d_in[8];
  const float* pe = (const float*)d_in[9];
  float* out = (float*)d_out;

  const long MT = (long)CB * CT;                 // 2048 rows
  char* ws = (char*)d_ws;
  size_t off = 0;
  auto alloc = [&](size_t bytes) -> void* {
    void* p = ws + off;
    off += (bytes + 255) & ~(size_t)255;
    return p;
  };
  h16*   xb  = (h16*)alloc(MT * CD * sizeof(h16));
  h16*   wqb = (h16*)alloc((long)CD * CD * sizeof(h16));
  h16*   wkb = (h16*)alloc((long)CD * CD * sizeof(h16));
  h16*   wvb = (h16*)alloc((long)CD * CD * sizeof(h16));
  h16*   wob = (h16*)alloc((long)CD * CD * sizeof(h16));
  h16*   peb = (h16*)alloc((long)CH * CMAXPOS * CDH * sizeof(h16));
  h16*   Qh  = (h16*)alloc(MT * CD * sizeof(h16));
  h16*   Kh  = (h16*)alloc(MT * CD * sizeof(h16));
  h16*   Vh  = (h16*)alloc(MT * CD * sizeof(h16));
  h16*   Vt  = (h16*)alloc(MT * CD * sizeof(h16));          // (B,H,DH,T)
  float* qk  = (float*)alloc((long)CB * CH * CT * CT * sizeof(float));
  float* qpe = (float*)alloc((long)CB * CH * CT * CMAXPOS * sizeof(float));
  h16*   atn = (h16*)alloc((long)CB * CH * CT * CT * sizeof(h16));
  h16*   Oh  = (h16*)alloc(MT * CD * sizeof(h16));          // attn*V, f16

  // 1) converts
  {
    long n = MT * CD;
    cvt_f32_f16<<<dim3((n + 1023) / 1024), 1024, 0, stream>>>(x, xb, n);
    n = (long)CD * CD;
    cvt_f32_f16<<<dim3((n + 1023) / 1024), 1024, 0, stream>>>(Wq, wqb, n);
    cvt_f32_f16<<<dim3((n + 1023) / 1024), 1024, 0, stream>>>(Wk, wkb, n);
    cvt_f32_f16<<<dim3((n + 1023) / 1024), 1024, 0, stream>>>(Wv, wvb, n);
    cvt_f32_f16<<<dim3((n + 1023) / 1024), 1024, 0, stream>>>(Wo, wob, n);
    n = (long)CH * CMAXPOS * CDH;
    cvt_f32_f16<<<dim3((n + 1023) / 1024), 1024, 0, stream>>>(pe, peb, n);
  }

  // 2) Q/K/V projections: (2048x1024) = xb * W^T   (f16 outputs)
  dim3 gProj(MT / 64, CD / 64, 1);
  gemm_wmma_f16<<<gProj, 128, 0, stream>>>(
      xb, 0, 0, CD,  wqb, 0, 0, CD,
      nullptr, 0, 0, 0,  Qh, 0, 0, CD,  bq, 1, (int)MT, CD, CD);
  gemm_wmma_f16<<<gProj, 128, 0, stream>>>(
      xb, 0, 0, CD,  wkb, 0, 0, CD,
      nullptr, 0, 0, 0,  Kh, 0, 0, CD,  bk, 1, (int)MT, CD, CD);
  gemm_wmma_f16<<<gProj, 128, 0, stream>>>(
      xb, 0, 0, CD,  wvb, 0, 0, CD,
      nullptr, 0, 0, 0,  Vh, 0, 0, CD,  bv, 1, (int)MT, CD, CD);

  // 2b) V -> (B,H,DH,T) so attn*V uses the vectorized TRANS_B path
  transpose_V<<<dim3(CT / 32, CD / 32, CB), dim3(32, 32), 0, stream>>>(Vh, Vt);

  // 3) qk[b,h] = Q_bh (TxDH) * K_bh^T  -> f32 (B,H,T,T)
  dim3 gQK(CT / 64, CT / 64, CB * CH);
  gemm_wmma_f16<<<gQK, 128, 0, stream>>>(
      Qh, (long)CT * CD, CDH, CD,
      Kh, (long)CT * CD, CDH, CD,
      qk, (long)CH * CT * CT, (long)CT * CT, CT,
      nullptr, 0, 0, 0,  nullptr, CH, CT, CT, CDH);

  // 4) qpe_tab[b,h] = Q_bh (TxDH) * pos_emb[h]^T -> f32 (B,H,T,64)
  dim3 gPE(CT / 64, CMAXPOS / 64, CB * CH);
  gemm_wmma_f16<<<gPE, 128, 0, stream>>>(
      Qh,  (long)CT * CD, CDH, CD,
      peb, 0, (long)CMAXPOS * CDH, CDH,
      qpe, (long)CH * CT * CMAXPOS, (long)CT * CMAXPOS, CMAXPOS,
      nullptr, 0, 0, 0,  nullptr, CH, CT, CMAXPOS, CDH);

  // 5) CoPE gates + cumsum + interp + causal softmax -> f16 attn
  cope_softmax<<<dim3(CT, CB * CH), 32, 0, stream>>>(qk, qpe, atn);

  // 6) out_bh = attn (TxT) * Vt_bh^T (64xT) -> f16 directly into (B,T,H*DH)
  dim3 gOV(CT / 64, CDH / 64, CB * CH);
  gemm_wmma_f16<<<gOV, 128, 0, stream>>>(
      atn, (long)CH * CT * CT, (long)CT * CT, CT,
      Vt,  (long)CH * CDH * CT, (long)CDH * CT, CT,
      nullptr, 0, 0, 0,
      Oh,  (long)CT * CD, CDH, CD,
      nullptr, CH, CT, CDH, CT);

  // 7) final projection: out = O * Wo^T + bo  (f32 output)
  gemm_wmma_f16<<<gProj, 128, 0, stream>>>(
      Oh, 0, 0, CD,  wob, 0, 0, CD,
      out, 0, 0, CD,  nullptr, 0, 0, 0,  bo, 1, (int)MT, CD, CD);
}
